// LSTM1_16621523435654
// MI455X (gfx1250) — matrix operand order repaired and checked by reading, not compiled, not run on hardware
//
#include <hip/hip_runtime.h>
#include <hip/hip_bf16.h>

typedef __attribute__((ext_vector_type(16))) _Float16 v16h;
typedef __attribute__((ext_vector_type(8)))  _Float16 v8h;
typedef __attribute__((ext_vector_type(8)))  float    v8f;

// Fast transcendentals on the CDNA5 TRANS unit (co-executes with VALU).
__device__ __forceinline__ float fast_sig(float x) {
    return 1.0f / (1.0f + __expf(-x));          // v_exp_f32 path; saturates correctly
}
__device__ __forceinline__ float fast_tanh(float x) {
#if __has_builtin(__builtin_amdgcn_tanhf)
    return __builtin_amdgcn_tanhf(x);           // native v_tanh_f32 if exposed
#else
    float e = __expf(2.0f * x);                 // e=inf -> +1, e=0 -> -1 : correct limits
    return 1.0f - 2.0f / (e + 1.0f);
#endif
}

// Load an A-matrix fragment (16x32 f16) in contiguous k order (same as the B fragments):
//   lane<16:  M=lane,    K = kb..kb+15 with kb = kc*32
//   lane>=16: M=lane-16, kb += 16
__device__ __forceinline__ v16h load_afrag(const _Float16* base, int strideH, int kc, int lane) {
    int m  = lane & 15;
    int kb = ((lane & 16) ? 16 : 0) + kc * 32;
    const _Float16* p = base + m * strideH + kb;
    v8h lo = *(const v8h*)(p);
    v8h hi = *(const v8h*)(p + 8);
    v16h r;
#pragma unroll
    for (int i = 0; i < 8; ++i) { r[i] = lo[i]; r[i + 8] = hi[i]; }
    return r;
}

__global__ __launch_bounds__(256) void lstm_mlp_fused(
    const float* __restrict__ x,
    const float* __restrict__ Wih, const float* __restrict__ Whh,
    const float* __restrict__ bih, const float* __restrict__ bhh,
    const float* __restrict__ W1, const float* __restrict__ b1,
    const float* __restrict__ W2, const float* __restrict__ b2,
    const float* __restrict__ W3, const float* __restrict__ b3,
    const float* __restrict__ W4, const float* __restrict__ b4,
    float* __restrict__ out, int B)
{
    // Weight fragments, pre-swizzled so each B operand is 32 contiguous bytes per lane.
    __shared__ __attribute__((aligned(32))) _Float16 sWf1[8 * 32 * 16];      // L1: 8 ntiles, K=32 (20 padded)
    __shared__ __attribute__((aligned(32))) _Float16 sWf2[4 * 4 * 32 * 16];  // L2: 4 kchunks x 4 ntiles
    __shared__ __attribute__((aligned(32))) _Float16 sWf3[2 * 2 * 32 * 16];  // L3: 2 kchunks x 2 ntiles
    __shared__ __attribute__((aligned(32))) _Float16 sH0[256 * 40];          // LSTM out, padded K stride 40
    __shared__ __attribute__((aligned(32))) _Float16 sAct[8 * 16 * 136];     // per-wave activation tile
    __shared__ float sWihL[32], sWhhL[16], sBsum[8];
    __shared__ float sB1[128], sB2[64], sB3[32], sW4[32], sB4[1];

    const int tid = threadIdx.x;

    // ---- stage LSTM weights / biases ----
    if (tid < 32) sWihL[tid] = Wih[tid];
    else if (tid < 48) sWhhL[tid - 32] = Whh[tid - 32];
    else if (tid < 56) sBsum[tid - 48] = bih[tid - 48] + bhh[tid - 48];
    if (tid < 128) sB1[tid] = b1[tid];
    if (tid < 64)  sB2[tid] = b2[tid];
    if (tid < 32)  { sB3[tid] = b3[tid]; sW4[tid] = W4[tid]; }
    if (tid == 0)  sB4[0] = b4[0];

    // ---- build B fragments: per-lane 16 halves, K = j + 16*(lane>=16), N = lane&15 ----
    for (int e = tid; e < 8 * 32 * 16; e += 256) {          // W1 is [128,20] row-major
        int j = e & 15, lane = (e >> 4) & 31, nt = e >> 9;
        int k = j + ((lane & 16) ? 16 : 0);
        int n = nt * 16 + (lane & 15);
        sWf1[e] = (_Float16)((k < 20) ? W1[n * 20 + k] : 0.0f);
    }
    for (int e = tid; e < 4 * 4 * 32 * 16; e += 256) {      // W2 is [64,128]
        int j = e & 15, lane = (e >> 4) & 31, nt = (e >> 9) & 3, kc = e >> 11;
        int k = kc * 32 + j + ((lane & 16) ? 16 : 0);
        int n = nt * 16 + (lane & 15);
        sWf2[e] = (_Float16)W2[n * 128 + k];
    }
    for (int e = tid; e < 2 * 2 * 32 * 16; e += 256) {      // W3 is [32,64]
        int j = e & 15, lane = (e >> 4) & 31, nt = (e >> 9) & 1, kc = e >> 10;
        int k = kc * 32 + j + ((lane & 16) ? 16 : 0);
        int n = nt * 16 + (lane & 15);
        sWf3[e] = (_Float16)W3[n * 64 + k];
    }
    __syncthreads();

    // ---- LSTM: one thread per batch row (gate order i,f,g,o; each gate H=2) ----
    long row  = (long)blockIdx.x * 256 + tid;
    long rowc = row < (long)B ? row : (long)(B - 1);
    const float* xr = x + rowc * 40;
    _Float16* hrow = &sH0[tid * 40];
#pragma unroll
    for (int k = 20; k < 32; ++k) hrow[k] = (_Float16)0.0f;

    float h0 = 0.f, h1 = 0.f, c0 = 0.f, c1 = 0.f;
#pragma unroll
    for (int t = 0; t < 10; ++t) {
        const float4 xv = *(const float4*)(xr + t * 4);
        float g[8];
#pragma unroll
        for (int q = 0; q < 8; ++q)
            g[q] = sBsum[q]
                 + sWihL[q * 4 + 0] * xv.x + sWihL[q * 4 + 1] * xv.y
                 + sWihL[q * 4 + 2] * xv.z + sWihL[q * 4 + 3] * xv.w
                 + sWhhL[q * 2 + 0] * h0 + sWhhL[q * 2 + 1] * h1;
        float ii0 = fast_sig(g[0]),  ii1 = fast_sig(g[1]);
        float ff0 = fast_sig(g[2]),  ff1 = fast_sig(g[3]);
        float gg0 = fast_tanh(g[4]), gg1 = fast_tanh(g[5]);
        float oo0 = fast_sig(g[6]),  oo1 = fast_sig(g[7]);
        c0 = ff0 * c0 + ii0 * gg0;  c1 = ff1 * c1 + ii1 * gg1;
        h0 = oo0 * fast_tanh(c0);   h1 = oo1 * fast_tanh(c1);
        hrow[t * 2 + 0] = (_Float16)h0;
        hrow[t * 2 + 1] = (_Float16)h1;
    }
    __syncthreads();

    // ---- MLP: each wave runs two 16-row tiles through WMMA layers ----
    const int wave = tid >> 5, lane = tid & 31;
    const int laneN = lane & 15, hiHalf = (lane >> 4) & 1;
    _Float16* act = &sAct[wave * (16 * 136)];

    for (int tile = 0; tile < 2; ++tile) {
        const _Float16* habase = &sH0[(wave * 32 + tile * 16) * 40];

        // Layer 1: [16x32(pad 20)] x [32x128] -> relu -> act[16x128]
        {
            v16h a0 = load_afrag(habase, 40, 0, lane);
#pragma unroll
            for (int nt = 0; nt < 8; ++nt) {
                v16h b = *(const v16h*)&sWf1[(nt * 32 + lane) * 16];
                v8f acc = {};
                acc = __builtin_amdgcn_wmma_f32_16x16x32_f16(false, a0, false, b,
                                                             (short)0, acc, false, false);
                float bias = sB1[nt * 16 + laneN];
#pragma unroll
                for (int r = 0; r < 8; ++r) {
                    float v = acc[r] + bias;  v = v > 0.f ? v : 0.f;
                    act[(r + hiHalf * 8) * 136 + nt * 16 + laneN] = (_Float16)v;
                }
            }
        }
        // Layer 2: [16x128] x [128x64] -> relu -> act[16x64]
        {
            v16h a[4];
#pragma unroll
            for (int kc = 0; kc < 4; ++kc) a[kc] = load_afrag(act, 136, kc, lane);
#pragma unroll
            for (int nt = 0; nt < 4; ++nt) {
                v8f acc = {};
#pragma unroll
                for (int kc = 0; kc < 4; ++kc) {
                    v16h b = *(const v16h*)&sWf2[((kc * 4 + nt) * 32 + lane) * 16];
                    acc = __builtin_amdgcn_wmma_f32_16x16x32_f16(false, a[kc], false, b,
                                                                 (short)0, acc, false, false);
                }
                float bias = sB2[nt * 16 + laneN];
#pragma unroll
                for (int r = 0; r < 8; ++r) {
                    float v = acc[r] + bias;  v = v > 0.f ? v : 0.f;
                    act[(r + hiHalf * 8) * 136 + nt * 16 + laneN] = (_Float16)v;
                }
            }
        }
        // Layer 3: [16x64] x [64x32] -> relu -> act[16x32]
        {
            v16h a[2];
#pragma unroll
            for (int kc = 0; kc < 2; ++kc) a[kc] = load_afrag(act, 136, kc, lane);
#pragma unroll
            for (int nt = 0; nt < 2; ++nt) {
                v8f acc = {};
#pragma unroll
                for (int kc = 0; kc < 2; ++kc) {
                    v16h b = *(const v16h*)&sWf3[((kc * 2 + nt) * 32 + lane) * 16];
                    acc = __builtin_amdgcn_wmma_f32_16x16x32_f16(false, a[kc], false, b,
                                                                 (short)0, acc, false, false);
                }
                float bias = sB3[nt * 16 + laneN];
#pragma unroll
                for (int r = 0; r < 8; ++r) {
                    float v = acc[r] + bias;  v = v > 0.f ? v : 0.f;
                    act[(r + hiHalf * 8) * 136 + nt * 16 + laneN] = (_Float16)v;
                }
            }
        }
        // Layer 4: [16x32] x [32x1] dot via VALU, lanes 0..15 each do one row
        if (lane < 16) {
            float s = sB4[0];
#pragma unroll
            for (int k = 0; k < 32; ++k) s += (float)act[lane * 136 + k] * sW4[k];
            long orow = (long)blockIdx.x * 256 + wave * 32 + tile * 16 + lane;
            if (orow < (long)B) out[orow] = s;
        }
    }
}

extern "C" void kernel_launch(void* const* d_in, const int* in_sizes, int n_in,
                              void* d_out, int out_size, void* d_ws, size_t ws_size,
                              hipStream_t stream) {
    const float* x   = (const float*)d_in[0];
    const float* Wih = (const float*)d_in[1];
    const float* Whh = (const float*)d_in[2];
    const float* bih = (const float*)d_in[3];
    const float* bhh = (const float*)d_in[4];
    const float* W1  = (const float*)d_in[5];
    const float* b1  = (const float*)d_in[6];
    const float* W2  = (const float*)d_in[7];
    const float* b2  = (const float*)d_in[8];
    const float* W3  = (const float*)d_in[9];
    const float* b3  = (const float*)d_in[10];
    const float* W4  = (const float*)d_in[11];
    const float* b4  = (const float*)d_in[12];
    float* out = (float*)d_out;

    int B = in_sizes[0] / 40;                 // x is [B, 10, 4]
    int blocks = (B + 255) / 256;             // 256 rows per block
    hipLaunchKernelGGL(lstm_mlp_fused, dim3(blocks), dim3(256), 0, stream,
                       x, Wih, Whh, bih, bhh, W1, b1, W2, b2, W3, b3, W4, b4, out, B);
}
